// DiMBlock_67525475827719
// MI455X (gfx1250) — matrix operand order repaired and checked by reading, not compiled, not run on hardware
//
#include <hip/hip_runtime.h>
#include <math.h>

// ---------------------------------------------------------------------------
// DiM block (adaLN MHA + adaLN GELU-FFN) for MI455X / gfx1250, wave32 + WMMA.
// All GEMMs: D = A(bf16, MxK row-major) x Bt(bf16, N-major: Bt[n][k]) with f32
// accumulation via v_wmma_f32_16x16x32_bf16. Tiles are prefetched into
// double-buffered LDS with GLOBAL_LOAD_ASYNC_TO_LDS_B128 (ASYNCcnt) when the
// toolchain exposes the builtin; otherwise a synchronous copy path is used.
// ---------------------------------------------------------------------------

typedef __bf16 bf16_t;
typedef __attribute__((ext_vector_type(16))) __bf16 v16bf;
typedef __attribute__((ext_vector_type(8)))  __bf16 v8bf;
typedef __attribute__((ext_vector_type(8)))  float  v8f;

#define D_    1024
#define H_    8
#define DH_   128
#define DFF_  4096
#define B_    8
#define N_    1024

#define LSTR 56   // LDS row stride in bf16 (112B: 16B aligned, conflict-free 16-row reads)

enum { MODE_BF16 = 0, MODE_F32SCALE = 1, MODE_GELU = 2, MODE_RESID = 3 };

#if defined(__gfx1250__) && __has_builtin(__builtin_amdgcn_global_load_async_to_lds_b128)
#define HAS_ASYNC_LDS 1
// builtin wants clang-native int vectors, NOT HIP_vector_type<int,4>
typedef int nv4i __attribute__((__vector_size__(4 * sizeof(int))));
typedef __attribute__((address_space(1))) nv4i g1_v4i;   // global b128
typedef __attribute__((address_space(3))) nv4i lds_v4i;  // LDS b128
// imm offset applies to BOTH global and LDS addresses (ISA §15.18 async pseudocode)
#define ASYNC_CP16(gp, lp, off) \
    __builtin_amdgcn_global_load_async_to_lds_b128((g1_v4i*)(gp), (lds_v4i*)(lp), (off), 0)
#else
#define HAS_ASYNC_LDS 0
#endif

static __device__ __forceinline__ void wait_async0() {
#if HAS_ASYNC_LDS
#if __has_builtin(__builtin_amdgcn_s_wait_asynccnt)
    __builtin_amdgcn_s_wait_asynccnt(0);
#else
    asm volatile("s_wait_asynccnt 0x0" ::: "memory");
#endif
#endif
}

// ---------------------------------------------------------------------------
// Tiled WMMA GEMM: block computes 128x128 of C; 8 waves, each 64(M)x32(N),
// i.e. 4x2 fragments of 16x16; K stepped by 32 (one WMMA K per step).
// Double-buffered LDS, async prefetch of the next K-tile overlaps compute.
// ---------------------------------------------------------------------------
template <int MODE>
__global__ __launch_bounds__(256) void gemm_wmma(
    const bf16_t* __restrict__ A,  int lda, long strideA,
    const bf16_t* __restrict__ Bt, int ldb, long strideB,
    void* __restrict__ Cout, int ldc, long strideC,
    const float* __restrict__ bias, int K, float alpha,
    const float* __restrict__ resid, const float* __restrict__ gate,
    int gateStride, int seqN)
{
    __shared__ bf16_t As[2][128 * LSTR];
    __shared__ bf16_t Bs[2][128 * LSTR];

    const int tid  = threadIdx.x;
    const int lane = tid & 31;
    const int wave = tid >> 5;
    const int wm   = wave & 1;       // 0..1 : 64-row slab
    const int wn   = wave >> 1;      // 0..3 : 32-col slab
    const int l16  = lane & 15;
    const int h8   = lane >> 4;      // half-wave selector (K subgroup)
    const int m0   = blockIdx.y * 128;
    const int n0   = blockIdx.x * 128;
    const int z    = blockIdx.z;

    A  += (long)z * strideA;
    Bt += (long)z * strideB;

    v8f acc[4][2];
#pragma unroll
    for (int r = 0; r < 4; ++r)
#pragma unroll
        for (int j = 0; j < 2; ++j)
#pragma unroll
            for (int e = 0; e < 8; ++e) acc[r][j][e] = 0.0f;

    const int rowL = tid >> 1;           // 0..127 tile row per thread
    const int colh = (tid & 1) * 16;     // 0 or 16 (bf16 elems)

    // per-thread 32B slice of the A/B tile copy
    auto load_tiles = [&](int buf, int k0) {
        const bf16_t* ga = A  + (long)(m0 + rowL) * lda + k0 + colh;
        const bf16_t* gb = Bt + (long)(n0 + rowL) * ldb + k0 + colh;
        bf16_t* la = &As[buf][rowL * LSTR + colh];
        bf16_t* lb = &Bs[buf][rowL * LSTR + colh];
#if HAS_ASYNC_LDS
        ASYNC_CP16(ga, la, 0);
        ASYNC_CP16(ga, la, 16);
        ASYNC_CP16(gb, lb, 0);
        ASYNC_CP16(gb, lb, 16);
#else
        const uint4* s0 = (const uint4*)ga;
        const uint4* s1 = (const uint4*)gb;
        uint4* d0 = (uint4*)la;
        uint4* d1 = (uint4*)lb;
        d0[0] = s0[0]; d0[1] = s0[1];
        d1[0] = s1[0]; d1[1] = s1[1];
#endif
    };

    load_tiles(0, 0);
    int buf = 0;

    for (int k0 = 0; k0 < K; k0 += 32) {
        wait_async0();        // this wave's tile copy into As/Bs[buf] landed
        __syncthreads();      // all waves' copies landed; prev reads of buf^1 done

        if (k0 + 32 < K) load_tiles(buf ^ 1, k0 + 32);   // overlaps compute below

        // B fragments: lane holds col n=l16, elems e -> K = 16*h8 + e (contiguous 32B)
        v16bf bfrag[2];
#pragma unroll
        for (int j = 0; j < 2; ++j) {
            const bf16_t* p = &Bs[buf][(wn * 32 + j * 16 + l16) * LSTR + h8 * 16];
            union { v16bf v; v8bf h[2]; } u;
            u.h[0] = *(const v8bf*)(p);
            u.h[1] = *(const v8bf*)(p + 8);
            bfrag[j] = u.v;
        }
        // A fragments: lane holds row m=l16, elems e -> K = 16*h8 + e
        // (two contiguous 16B pieces, matching B)
#pragma unroll
        for (int r = 0; r < 4; ++r) {
            const bf16_t* p = &As[buf][(wm * 64 + r * 16 + l16) * LSTR + h8 * 16];
            union { v16bf v; v8bf h[2]; } u;
            u.h[0] = *(const v8bf*)(p);
            u.h[1] = *(const v8bf*)(p + 8);
            v16bf afrag = u.v;
#pragma unroll
            for (int j = 0; j < 2; ++j)
                acc[r][j] = __builtin_amdgcn_wmma_f32_16x16x32_bf16(
                    false, afrag, false, bfrag[j], (short)0, acc[r][j], false, false);
        }
        buf ^= 1;
    }

    // Epilogue. C layout: lanes 0-15 N=lane, vgpr v -> M=v; lanes 16-31 M=8+v.
#pragma unroll
    for (int r = 0; r < 4; ++r) {
#pragma unroll
        for (int j = 0; j < 2; ++j) {
            const int nn = n0 + wn * 32 + j * 16 + l16;
            const int mb = m0 + wm * 64 + r * 16 + h8 * 8;
            float bv = 0.0f;
            if (MODE != MODE_F32SCALE && bias) bv = bias[nn];
#pragma unroll
            for (int v = 0; v < 8; ++v) {
                const long row = mb + v;
                const float val = acc[r][j][v];
                if (MODE == MODE_BF16) {
                    ((bf16_t*)Cout)[(long)z * strideC + row * ldc + nn] = (bf16_t)(val + bv);
                } else if (MODE == MODE_F32SCALE) {
                    ((float*)Cout)[(long)z * strideC + row * ldc + nn] = val * alpha;
                } else if (MODE == MODE_GELU) {
                    const float t = val + bv;
                    const float g = 0.5f * t * (1.0f + erff(t * 0.70710678118f));
                    ((bf16_t*)Cout)[(long)z * strideC + row * ldc + nn] = (bf16_t)g;
                } else { // MODE_RESID: out = resid + gate[b]*(acc+bias), fp32
                    const int b = (int)(row / seqN);
                    const float t = val + bv;
                    ((float*)Cout)[row * ldc + nn] =
                        resid[row * ldc + nn] + gate[(long)b * gateStride + nn] * t;
                }
            }
        }
    }
}

// ---------------------------------------------------------------------------
// LayerNorm + adaLN modulate -> bf16 A-matrix. One block per (b,n) row.
// ---------------------------------------------------------------------------
__global__ __launch_bounds__(256) void ln_mod_kernel(
    const float* __restrict__ x, const float* __restrict__ g,
    const float* __restrict__ bta, const float* __restrict__ mod,
    bf16_t* __restrict__ out, int Dd, int seqN)
{
    const long row = blockIdx.x;
    const int  b   = (int)(row / seqN);
    const float* xr = x + row * Dd;
    __shared__ float red[256];

    float s = 0.0f;
    for (int d = threadIdx.x; d < Dd; d += 256) s += xr[d];
    red[threadIdx.x] = s; __syncthreads();
    for (int off = 128; off > 0; off >>= 1) {
        if (threadIdx.x < off) red[threadIdx.x] += red[threadIdx.x + off];
        __syncthreads();
    }
    const float mu = red[0] / Dd;
    __syncthreads();

    float s2 = 0.0f;
    for (int d = threadIdx.x; d < Dd; d += 256) { float t = xr[d] - mu; s2 += t * t; }
    red[threadIdx.x] = s2; __syncthreads();
    for (int off = 128; off > 0; off >>= 1) {
        if (threadIdx.x < off) red[threadIdx.x] += red[threadIdx.x + off];
        __syncthreads();
    }
    const float rstd = rsqrtf(red[0] / Dd + 1e-6f);

    const float* mrow = mod + (long)b * 3 * Dd;   // [shift | scale | gate]
    bf16_t* orow = out + row * Dd;
    for (int d = threadIdx.x; d < Dd; d += 256) {
        float hv = (xr[d] - mu) * rstd * g[d] + bta[d];
        hv = hv * (1.0f + mrow[Dd + d]) + mrow[d];
        orow[d] = (bf16_t)hv;
    }
}

// ---------------------------------------------------------------------------
// Row softmax: f32 scores -> bf16 probs. One block per row.
// ---------------------------------------------------------------------------
__global__ __launch_bounds__(256) void softmax_kernel(
    const float* __restrict__ S, bf16_t* __restrict__ P, int W)
{
    const long row = blockIdx.x;
    const float* sr = S + row * W;
    bf16_t* pr = P + row * W;
    __shared__ float red[256];

    float m = -1e30f;
    for (int i = threadIdx.x; i < W; i += 256) m = fmaxf(m, sr[i]);
    red[threadIdx.x] = m; __syncthreads();
    for (int off = 128; off > 0; off >>= 1) {
        if (threadIdx.x < off) red[threadIdx.x] = fmaxf(red[threadIdx.x], red[threadIdx.x + off]);
        __syncthreads();
    }
    m = red[0]; __syncthreads();

    float s = 0.0f;
    for (int i = threadIdx.x; i < W; i += 256) s += __expf(sr[i] - m);
    red[threadIdx.x] = s; __syncthreads();
    for (int off = 128; off > 0; off >>= 1) {
        if (threadIdx.x < off) red[threadIdx.x] += red[threadIdx.x + off];
        __syncthreads();
    }
    const float inv = 1.0f / red[0];
    for (int i = threadIdx.x; i < W; i += 256)
        pr[i] = (bf16_t)(__expf(sr[i] - m) * inv);
}

// ---------------------------------------------------------------------------
// mod = silu(c) @ W + bias   (small, fp32 VALU; W is (D,3D) K-major)
// ---------------------------------------------------------------------------
__global__ __launch_bounds__(256) void ada_kernel(
    const float* __restrict__ c, const float* __restrict__ W,
    const float* __restrict__ bias, float* __restrict__ out, int Dd, int N3)
{
    const int b = blockIdx.y;
    const int col = blockIdx.x * 256 + threadIdx.x;
    __shared__ float sc[D_];
    for (int k = threadIdx.x; k < Dd; k += 256) {
        const float v = c[(long)b * Dd + k];
        sc[k] = v / (1.0f + __expf(-v));
    }
    __syncthreads();
    float acc = 0.0f;
    for (int k = 0; k < Dd; ++k) acc += sc[k] * W[(long)k * N3 + col];
    out[(long)b * N3 + col] = acc + bias[col];
}

// fp32 -> bf16 (same layout)
__global__ void conv_bf16(const float* __restrict__ s, bf16_t* __restrict__ d, long n) {
    const long i = (long)blockIdx.x * 256 + threadIdx.x;
    if (i < n) d[i] = (bf16_t)s[i];
}
// fp32 (R,C) -> bf16 (C,R)  (make weight N-major)
__global__ void conv_bf16_t(const float* __restrict__ s, bf16_t* __restrict__ d, int R, int C) {
    const long i = (long)blockIdx.x * 256 + threadIdx.x;
    if (i < (long)R * C) {
        const int r = (int)(i / C), c = (int)(i % C);
        d[(long)c * R + r] = (bf16_t)s[i];
    }
}

// qkv (B,N,3D) bf16 -> q,k (B,H,N,DH) and vT (B,H,DH,N)
__global__ void split_qkv(const bf16_t* __restrict__ qkv,
                          bf16_t* __restrict__ q, bf16_t* __restrict__ k,
                          bf16_t* __restrict__ vT)
{
    const long i = (long)blockIdx.x * 256 + threadIdx.x;
    const long total = (long)B_ * N_ * D_;
    if (i >= total) return;
    const int d = (int)(i % D_);
    const long t = i / D_;
    const int n = (int)(t % N_);
    const int b = (int)(t / N_);
    const int h = d / DH_, dd = d % DH_;
    const bf16_t* src = qkv + ((long)b * N_ + n) * (3 * D_);
    const long qidx = (((long)b * H_ + h) * N_ + n) * DH_ + dd;
    q[qidx] = src[d];
    k[qidx] = src[D_ + d];
    vT[(((long)b * H_ + h) * DH_ + dd) * (long)N_ + n] = src[2 * D_ + d];
}

// ---------------------------------------------------------------------------
extern "C" void kernel_launch(void* const* d_in, const int* in_sizes, int n_in,
                              void* d_out, int out_size, void* d_ws, size_t ws_size,
                              hipStream_t stream)
{
    const float* x          = (const float*)d_in[0];
    const float* c          = (const float*)d_in[1];
    const float* m_ada_w    = (const float*)d_in[2];
    const float* m_ada_b    = (const float*)d_in[3];
    const float* m_norm_g   = (const float*)d_in[4];
    const float* m_norm_b   = (const float*)d_in[5];
    const float* in_proj_w  = (const float*)d_in[6];
    const float* in_proj_b  = (const float*)d_in[7];
    const float* out_proj_w = (const float*)d_in[8];
    const float* out_proj_b = (const float*)d_in[9];
    const float* f_ada_w    = (const float*)d_in[10];
    const float* f_ada_b    = (const float*)d_in[11];
    const float* f_norm_g   = (const float*)d_in[12];
    const float* f_norm_b   = (const float*)d_in[13];
    const float* w1         = (const float*)d_in[14];
    const float* b1         = (const float*)d_in[15];
    const float* w2         = (const float*)d_in[16];
    const float* b2         = (const float*)d_in[17];
    float* out = (float*)d_out;

    // ---- workspace carve-up (256B aligned) ----
    char* base = (char*)d_ws;
    size_t off = 0;
    auto carve = [&](size_t bytes) -> char* {
        char* p = base + off;
        off = (off + bytes + 255) & ~(size_t)255;
        return p;
    };
    bf16_t* w_inp  = (bf16_t*)carve((size_t)3 * D_ * D_ * 2);        // in_proj  (3D,D) N-major
    bf16_t* w_outp = (bf16_t*)carve((size_t)D_ * D_ * 2);            // out_proj (D,D)  N-major
    bf16_t* w1T    = (bf16_t*)carve((size_t)DFF_ * D_ * 2);          // (DFF,D)
    bf16_t* w2T    = (bf16_t*)carve((size_t)D_ * DFF_ * 2);          // (D,DFF)
    float*  mod    = (float*)carve((size_t)B_ * 3 * D_ * 4);
    float*  mod2   = (float*)carve((size_t)B_ * 3 * D_ * 4);
    bf16_t* hbuf   = (bf16_t*)carve((size_t)B_ * N_ * D_ * 2);       // modulated LN (bf16)
    bf16_t* qb     = (bf16_t*)carve((size_t)B_ * H_ * N_ * DH_ * 2);
    bf16_t* kb     = (bf16_t*)carve((size_t)B_ * H_ * N_ * DH_ * 2);
    bf16_t* vTb    = (bf16_t*)carve((size_t)B_ * H_ * DH_ * N_ * 2);
    bf16_t* attnb  = (bf16_t*)carve((size_t)H_ * N_ * N_ * 2);       // one batch deep
    bf16_t* obuf   = (bf16_t*)carve((size_t)B_ * N_ * D_ * 2);
    float*  xmid   = (float*)carve((size_t)B_ * N_ * D_ * 4);
    bf16_t* qkvb   = (bf16_t*)carve((size_t)B_ * N_ * 3 * D_ * 2);
    float*  scores = (float*)carve((size_t)H_ * N_ * N_ * 4);        // one batch deep
    bf16_t* act    = (bf16_t*)qkvb;  // FFN hidden (B,N,DFF) reuses qkv+scores region

    const long M = (long)B_ * N_;       // 8192 token rows
    const float inv_sqrt_dh = 0.08838834764831845f; // 1/sqrt(128)

    // 1) weights -> bf16, N-major
    conv_bf16<<<dim3((3 * D_ * D_ + 255) / 256), 256, 0, stream>>>(in_proj_w, w_inp, (long)3 * D_ * D_);
    conv_bf16<<<dim3((D_ * D_ + 255) / 256), 256, 0, stream>>>(out_proj_w, w_outp, (long)D_ * D_);
    conv_bf16_t<<<dim3((D_ * DFF_ + 255) / 256), 256, 0, stream>>>(w1, w1T, D_, DFF_);
    conv_bf16_t<<<dim3((DFF_ * D_ + 255) / 256), 256, 0, stream>>>(w2, w2T, DFF_, D_);

    // 2) adaLN modulation vectors
    ada_kernel<<<dim3(3 * D_ / 256, B_), 256, 0, stream>>>(c, m_ada_w, m_ada_b, mod, D_, 3 * D_);
    ada_kernel<<<dim3(3 * D_ / 256, B_), 256, 0, stream>>>(c, f_ada_w, f_ada_b, mod2, D_, 3 * D_);

    // 3) LN + modulate -> h (bf16)
    ln_mod_kernel<<<dim3((unsigned)M), 256, 0, stream>>>(x, m_norm_g, m_norm_b, mod, hbuf, D_, N_);

    // 4) qkv = h @ in_projT + b  (8192 x 3072, K=1024) -> bf16
    gemm_wmma<MODE_BF16><<<dim3(3 * D_ / 128, M / 128, 1), 256, 0, stream>>>(
        hbuf, D_, 0, w_inp, D_, 0, qkvb, 3 * D_, 0, in_proj_b, D_, 1.0f,
        nullptr, nullptr, 0, N_);

    // 5) split into q,k (B,H,N,DH) and vT (B,H,DH,N)
    split_qkv<<<dim3((unsigned)((M * D_ + 255) / 256)), 256, 0, stream>>>(qkvb, qb, kb, vTb);

    // 6) attention, one batch at a time (stream-ordered; reuses scores/attnb)
    for (int b = 0; b < B_; ++b) {
        const long hoff = (long)b * H_ * N_ * DH_;
        // scores = q @ k^T / sqrt(dh)   (per head: 1024x1024, K=128) -> f32
        gemm_wmma<MODE_F32SCALE><<<dim3(N_ / 128, N_ / 128, H_), 256, 0, stream>>>(
            qb + hoff, DH_, (long)N_ * DH_, kb + hoff, DH_, (long)N_ * DH_,
            scores, N_, (long)N_ * N_, nullptr, DH_, inv_sqrt_dh,
            nullptr, nullptr, 0, N_);
        // softmax rows -> bf16 probs
        softmax_kernel<<<dim3(H_ * N_), 256, 0, stream>>>(scores, attnb, N_);
        // o = attn @ v  (per head: 1024x128, K=1024) -> obuf (B,N,H,DH) bf16
        gemm_wmma<MODE_BF16><<<dim3(1, N_ / 128, H_), 256, 0, stream>>>(
            attnb, N_, (long)N_ * N_, vTb + hoff, N_, (long)DH_ * N_,
            obuf + (long)b * N_ * D_, D_, DH_, nullptr, N_, 1.0f,
            nullptr, nullptr, 0, N_);
    }

    // 7) xmid = x + gate * (o @ out_projT + b)   (8192x1024, K=1024) -> f32
    gemm_wmma<MODE_RESID><<<dim3(D_ / 128, M / 128, 1), 256, 0, stream>>>(
        obuf, D_, 0, w_outp, D_, 0, xmid, D_, 0, out_proj_b, D_, 1.0f,
        x, mod + 2 * D_, 3 * D_, N_);

    // 8) LN + modulate (FFN branch) -> h2 (bf16, reuses hbuf)
    ln_mod_kernel<<<dim3((unsigned)M), 256, 0, stream>>>(xmid, f_norm_g, f_norm_b, mod2, hbuf, D_, N_);

    // 9) act = gelu(h2 @ w1 + b1)   (8192x4096, K=1024) -> bf16
    gemm_wmma<MODE_GELU><<<dim3(DFF_ / 128, M / 128, 1), 256, 0, stream>>>(
        hbuf, D_, 0, w1T, D_, 0, act, DFF_, 0, b1, D_, 1.0f,
        nullptr, nullptr, 0, N_);

    // 10) out = xmid + gate2 * (act @ w2 + b2)   (8192x1024, K=4096) -> f32
    gemm_wmma<MODE_RESID><<<dim3(D_ / 128, M / 128, 1), 256, 0, stream>>>(
        act, DFF_, 0, w2T, DFF_, 0, out, D_, 0, b2, DFF_, 1.0f,
        xmid, mod2 + 2 * D_, 3 * D_, N_);

    (void)in_sizes; (void)n_in; (void)out_size; (void)ws_size;
}
